// LAS_76587856822428
// MI455X (gfx1250) — compile-verified, matrix operand order checked by reading, not run on hardware
//
#include <hip/hip_runtime.h>
#include <hip/hip_bf16.h>
#include <stdint.h>

#define B_   64
#define T_   1024
#define U_   100
#define UM1  99
#define V_   5000
#define D_   256

typedef __attribute__((ext_vector_type(16))) __bf16 v16bf;
typedef __attribute__((ext_vector_type(8)))  float  v8f;
typedef unsigned short us;

union Frag16 { uint4 u[2]; v16bf v; };

__device__ __forceinline__ us f2bf(float f) {
    unsigned int u = __float_as_uint(f);
    u += 0x7FFFu + ((u >> 16) & 1u);          // round-to-nearest-even
    return (us)(u >> 16);
}
__device__ __forceinline__ float bf2f(us h) {
    return __uint_as_float(((unsigned int)h) << 16);
}
__device__ __forceinline__ float sigf(float x) { return 1.0f / (1.0f + __expf(-x)); }

// Load one lane's share of a 16x32 bf16 A/B fragment.
// p points at (row base + k_chunk); lane half (0/1) selects K {0..7,16..23} vs {8..15,24..31}.
__device__ __forceinline__ v16bf load_frag(const us* p, int half) {
    Frag16 f;
    f.u[0] = *(const uint4*)(p + half * 8);
    f.u[1] = *(const uint4*)(p + 16 + half * 8);
    return f.v;
}
__device__ __forceinline__ v16bf zero_frag() {
    Frag16 f;
    f.u[0] = make_uint4(0u, 0u, 0u, 0u);
    f.u[1] = make_uint4(0u, 0u, 0u, 0u);
    return f.v;
}
__device__ __forceinline__ v8f wmma_bf16(v16bf a, v16bf b, v8f c) {
    return __builtin_amdgcn_wmma_f32_16x16x32_bf16(false, a, false, b, (short)0, c, false, false);
}

// ---------------- prep kernels ----------------

__global__ void k_f2bf(const float* __restrict__ s, us* __restrict__ d, int n) {
    int i = blockIdx.x * blockDim.x + threadIdx.x;
    if (i < n) d[i] = f2bf(s[i]);
}
__global__ void k_zero_f(float* p, int n) {
    int i = blockIdx.x * blockDim.x + threadIdx.x;
    if (i < n) p[i] = 0.0f;
}
__global__ void k_zero_us(us* p, int n) {
    int i = blockIdx.x * blockDim.x + threadIdx.x;
    if (i < n) p[i] = (us)0;
}
// fused [Wih(1024x512) | Whh(1024x256)] -> bf16 (1024x768)
__global__ void k_catw(const float* __restrict__ wih, const float* __restrict__ whh,
                       us* __restrict__ dst) {
    int i = blockIdx.x * blockDim.x + threadIdx.x;
    if (i >= 1024 * 768) return;
    int n = i / 768, k = i % 768;
    float v = (k < 512) ? wih[n * 512 + k] : whh[n * 256 + (k - 512)];
    dst[i] = f2bf(v);
}
// v_w (D,D) -> transpose fp32 so state GEMV reads coalesced
__global__ void k_trvw(const float* __restrict__ vw, float* __restrict__ vwT) {
    int i = blockIdx.x * blockDim.x + threadIdx.x;
    if (i >= D_ * D_) return;
    int k = i / D_, d = i % D_;
    vwT[i] = vw[d * D_ + k];
}
// embedding gather: emb_bf16[b,t,d] = embedding[y[b,t], d], t in [0,99)
__global__ void k_gemb(const float* __restrict__ emb, const int* __restrict__ y,
                       us* __restrict__ out) {
    int i = blockIdx.x * blockDim.x + threadIdx.x;
    if (i >= B_ * UM1 * D_) return;
    int d = i & (D_ - 1);
    int bt = i >> 8;
    int b = bt / UM1, t = bt % UM1;
    int tok = y[b * U_ + t];
    out[i] = f2bf(emb[(long)tok * D_ + d]);
}
// xatt init: [e_0 | 0 | 0]
__global__ void k_initx(us* __restrict__ xatt, const us* __restrict__ embb) {
    int i = blockIdx.x * blockDim.x + threadIdx.x;
    if (i >= B_ * 768) return;
    int b = i / 768, k = i % 768;
    xatt[i] = (k < D_) ? embb[(b * UM1) * D_ + k] : (us)0;
}

// ---------------- att_h = bf16(eout @ w_w.T + w_b) ----------------
// grid (4096, 4), block 32. One wave computes a 16x64 tile, K=256.
// Fully unrolled; scheduler hoists next-chunk loads across WMMAs via renaming.
__global__ void k_atth(const us* __restrict__ eoutb, const us* __restrict__ ww,
                       const float* __restrict__ wb, us* __restrict__ out) {
    int mtile = blockIdx.x, ngrp = blockIdx.y;
    int lane = threadIdx.x & 31, r = lane & 15, half = lane >> 4;
    const us* arow = eoutb + (long)(mtile * 16 + r) * D_;
    v8f z = {};
    v8f acc[4] = {z, z, z, z};
#pragma unroll
    for (int kc = 0; kc < 8; ++kc) {
        int kb = kc * 32;
        v16bf a = load_frag(arow + kb, half);
#pragma unroll
        for (int nt = 0; nt < 4; ++nt) {
            int n = ngrp * 64 + nt * 16 + r;
            v16bf b = load_frag(ww + n * D_ + kb, half);
            acc[nt] = wmma_bf16(a, b, acc[nt]);
        }
    }
    int col = lane & 15;
#pragma unroll
    for (int nt = 0; nt < 4; ++nt) {
        int ng = ngrp * 64 + nt * 16 + col;
        float bias = wb[ng];
#pragma unroll
        for (int j = 0; j < 8; ++j) {
            long row = (long)mtile * 16 + half * 8 + j;
            out[row * D_ + ng] = f2bf(acc[nt][j] + bias);
        }
    }
}

// ---------------- fused LSTM step (gates GEMM + cell) ----------------
// grid (4,16), block 384 (12 waves = 4 gates x 3 K-splits of 256 each).
// Wave (gate,split) computes a 16x16 partial over its K region; partials are
// reduced in LDS and the cell nonlinearity is applied block-locally.
__global__ void k_lstm(const us* __restrict__ x0, long ld0,
                       const us* __restrict__ x1, long ld1,
                       const us* __restrict__ x2, long ld2,
                       const us* __restrict__ W,
                       const float* __restrict__ bih, const float* __restrict__ bhh,
                       float* __restrict__ cbuf,
                       float* __restrict__ hf32,
                       us* __restrict__ hun, long ldun,
                       us* __restrict__ hmask, long ldm,
                       const float* __restrict__ ymask, int s) {
    __shared__ float gl[12][16][16];
    int tid = threadIdx.x;
    int w = tid >> 5;                 // 0..11
    int gate = w / 3, split = w - gate * 3;
    int lane = tid & 31, r = lane & 15, half = lane >> 4;
    int mtile = blockIdx.x, dtile = blockIdx.y;
    int row = mtile * 16 + r;
    int n = gate * D_ + dtile * 16 + r;
    const us* xb; long ld;
    if (split == 0)      { xb = x0; ld = ld0; }
    else if (split == 1) { xb = x1; ld = ld1; }
    else                 { xb = x2; ld = ld2; }
    const us* arow = xb + row * ld;
    const us* brow = W + (long)n * 768 + split * 256;
    v8f acc = {};
#pragma unroll
    for (int j = 0; j < 8; ++j) {
        v16bf a = load_frag(arow + j * 32, half);
        v16bf b = load_frag(brow + j * 32, half);
        acc = wmma_bf16(a, b, acc);
    }
    int col = lane & 15;
#pragma unroll
    for (int j = 0; j < 8; ++j) gl[w][half * 8 + j][col] = acc[j];
    __syncthreads();
    if (tid < 256) {
        int rr = tid >> 4, cc = tid & 15;
        float g[4];
#pragma unroll
        for (int gt = 0; gt < 4; ++gt) {
            int ng = gt * D_ + dtile * 16 + cc;
            g[gt] = gl[gt * 3][rr][cc] + gl[gt * 3 + 1][rr][cc] + gl[gt * 3 + 2][rr][cc]
                  + bih[ng] + bhh[ng];
        }
        int b2 = mtile * 16 + rr, d = dtile * 16 + cc;
        float cold = cbuf[b2 * D_ + d];
        float cn = sigf(g[1]) * cold + sigf(g[0]) * tanhf(g[2]);
        float hn = sigf(g[3]) * tanhf(cn);
        cbuf[b2 * D_ + d] = cn;
        if (hf32) hf32[b2 * D_ + d] = hn;
        if (hun)  hun[b2 * ldun + d] = f2bf(hn);
        if (hmask) {
            float m = ymask[b2 * U_ + s + 1];
            hmask[b2 * ldm + d] = f2bf(hn * m);
        }
    }
}

// ---------------- attention step (state GEMV + score/softmax/context) ----------------
// grid 64 (one block per batch row), block 256.
__global__ void k_attn(const us* __restrict__ atthb, const us* __restrict__ eoutb,
                       const us* __restrict__ embb,
                       const float* __restrict__ h, const float* __restrict__ vwT,
                       const float* __restrict__ vb, const float* __restrict__ wav,
                       const float* __restrict__ xmask, const float* __restrict__ ymask,
                       us* __restrict__ xatt, us* __restrict__ ctxsb, int s) {
    __shared__ float st[D_];
    __shared__ float sc[T_];
    __shared__ float red[8];
    __shared__ float bmax, bsum;
    int tid = threadIdx.x, b = blockIdx.x;
    // state[d] = h[b,:] . v_w[d,:] + v_b[d]  (vwT coalesced)
    {
        float a = vb[tid];
        const float* hrow = h + b * D_;
        for (int k = 0; k < D_; ++k) a += hrow[k] * vwT[k * D_ + tid];
        st[tid] = a;
    }
    __syncthreads();
    int wid = tid >> 5, lane = tid & 31;
    int d0 = lane * 8;
    float wv[8], stl[8];
#pragma unroll
    for (int j = 0; j < 8; ++j) { wv[j] = wav[d0 + j]; stl[j] = st[d0 + j]; }
    // scores: 8 waves, one t per wave per iteration; lane covers 8 contiguous d
    for (int t = wid; t < T_; t += 8) {
        const us* ap = atthb + ((long)b * T_ + t) * D_ + d0;
        uint4 pk = *(const uint4*)ap;
        const us* e = (const us*)&pk;
        float p = 0.f;
#pragma unroll
        for (int j = 0; j < 8; ++j) p += tanhf(stl[j] + bf2f(e[j])) * wv[j];
#pragma unroll
        for (int o = 16; o > 0; o >>= 1) p += __shfl_xor(p, o, 32);
        if (lane == 0) sc[t] = p + (xmask[b * T_ + t] - 1.0f) * 1e30f;
    }
    __syncthreads();
    // softmax max
    float m = -3.0e38f;
    for (int t = tid; t < T_; t += 256) m = fmaxf(m, sc[t]);
#pragma unroll
    for (int o = 16; o > 0; o >>= 1) m = fmaxf(m, __shfl_xor(m, o, 32));
    if (lane == 0) red[wid] = m;
    __syncthreads();
    if (tid == 0) {
        float mm = red[0];
        for (int i = 1; i < 8; ++i) mm = fmaxf(mm, red[i]);
        bmax = mm;
    }
    __syncthreads();
    // softmax sum
    float ss = 0.f;
    for (int t = tid; t < T_; t += 256) {
        float e = __expf(sc[t] - bmax);
        sc[t] = e;
        ss += e;
    }
#pragma unroll
    for (int o = 16; o > 0; o >>= 1) ss += __shfl_xor(ss, o, 32);
    if (lane == 0) red[wid] = ss;
    __syncthreads();
    if (tid == 0) {
        float t2 = 0.f;
        for (int i = 0; i < 8; ++i) t2 += red[i];
        bsum = t2;
    }
    __syncthreads();
    float inv = 1.0f / bsum;
    // ctx[d] = sum_t a_t * eout[b,t,d]  (coalesced across d)
    float a = 0.f;
    for (int t = 0; t < T_; ++t) a += sc[t] * bf2f(eoutb[((long)b * T_ + t) * D_ + tid]);
    a *= inv;
    float ym = ymask[b * U_ + s + 1];
    ctxsb[((long)b * UM1 + s) * D_ + tid] = f2bf(a * ym);
    xatt[b * 768 + 256 + tid] = f2bf(a);
    xatt[b * 768 + 512 + tid] = f2bf(h[b * D_ + tid]);
    if (s + 1 < UM1) xatt[b * 768 + tid] = embb[((long)b * UM1 + s + 1) * D_ + tid];
}

// ---------------- classifier: logit = [lstms|ctxs|dech] @ cls_w.T + cls_b ----------------
// grid (198, 79), block 32. One wave -> 32x64 tile (2 M-subtiles share each B fragment).
// K=768 in 3 regions of 256. Fully unrolled; B fragments are consumed immediately.
__global__ void k_cls(const us* __restrict__ lst, const us* __restrict__ ctx,
                      const us* __restrict__ dec, const us* __restrict__ Wc,
                      const float* __restrict__ cb, float* __restrict__ out) {
    int mtile = blockIdx.x, ngrp = blockIdx.y;
    int lane = threadIdx.x & 31, r = lane & 15, half = lane >> 4;
    long row0 = (long)mtile * 32 + r;
    long row1 = row0 + 16;
    int nbase = ngrp * 64 + r;

    v8f z = {};
    v8f acc0[4] = {z, z, z, z};
    v8f acc1[4] = {z, z, z, z};

#pragma unroll
    for (int kc = 0; kc < 24; ++kc) {
        int kb = kc * 32;
        const us* ap0;
        const us* ap1;
        if (kb < 256)      { ap0 = lst + row0 * D_ + kb;         ap1 = lst + row1 * D_ + kb; }
        else if (kb < 512) { ap0 = ctx + row0 * D_ + (kb - 256); ap1 = ctx + row1 * D_ + (kb - 256); }
        else               { ap0 = dec + row0 * D_ + (kb - 512); ap1 = dec + row1 * D_ + (kb - 512); }
        v16bf a0 = load_frag(ap0, half);
        v16bf a1 = load_frag(ap1, half);
#pragma unroll
        for (int nt = 0; nt < 4; ++nt) {
            int n = nbase + nt * 16;
            v16bf b = (n < V_) ? load_frag(Wc + (long)n * 768 + kb, half) : zero_frag();
            acc0[nt] = wmma_bf16(a0, b, acc0[nt]);
            acc1[nt] = wmma_bf16(a1, b, acc1[nt]);
        }
    }

    int col = lane & 15;
#pragma unroll
    for (int nt = 0; nt < 4; ++nt) {
        int ng = ngrp * 64 + nt * 16 + col;
        if (ng >= V_) continue;
        float bias = cb[ng];
#pragma unroll
        for (int j = 0; j < 8; ++j) {
            long orow0 = (long)mtile * 32 + half * 8 + j;       // == b*99 + t
            out[orow0 * V_ + ng] = acc0[nt][j] + bias;
            out[(orow0 + 16) * V_ + ng] = acc1[nt][j] + bias;
        }
    }
}

// ---------------- host ----------------

extern "C" void kernel_launch(void* const* d_in, const int* in_sizes, int n_in,
                              void* d_out, int out_size, void* d_ws, size_t ws_size,
                              hipStream_t stream) {
    (void)in_sizes; (void)n_in; (void)out_size; (void)ws_size;
    const float* eout    = (const float*)d_in[0];
    const float* xmask   = (const float*)d_in[1];
    const int*   y       = (const int*)d_in[2];
    const float* ymask   = (const float*)d_in[3];
    const float* embedding = (const float*)d_in[4];
    const float* w_w     = (const float*)d_in[5];
    const float* w_b     = (const float*)d_in[6];
    const float* v_w     = (const float*)d_in[7];
    const float* v_b     = (const float*)d_in[8];
    const float* w_att_v = (const float*)d_in[9];
    const float* att_Wih = (const float*)d_in[10];
    const float* att_Whh = (const float*)d_in[11];
    const float* att_bih = (const float*)d_in[12];
    const float* att_bhh = (const float*)d_in[13];
    const float* dec_Wih = (const float*)d_in[14];
    const float* dec_Whh = (const float*)d_in[15];
    const float* dec_bih = (const float*)d_in[16];
    const float* dec_bhh = (const float*)d_in[17];
    const float* cls_w   = (const float*)d_in[18];
    const float* cls_b   = (const float*)d_in[19];
    float* out = (float*)d_out;

    char* base = (char*)d_ws;
    size_t off = 0;
    auto carve = [&](size_t bytes) -> char* {
        char* p = base + off;
        off = (off + bytes + 255) & ~(size_t)255;
        return p;
    };
    us*    eoutb = (us*)   carve((size_t)B_ * T_ * D_ * 2);       // eout bf16
    us*    atthb = (us*)   carve((size_t)B_ * T_ * D_ * 2);       // att_h bf16
    us*    attW  = (us*)   carve((size_t)1024 * 768 * 2);         // [att_Wih|att_Whh]
    us*    decW  = (us*)   carve((size_t)1024 * 768 * 2);         // [dec_Wih|dec_Whh]
    us*    wwb   = (us*)   carve((size_t)D_ * D_ * 2);            // w_w bf16
    us*    clsW  = (us*)   carve((size_t)V_ * 768 * 2);           // cls_w bf16
    float* vwT   = (float*)carve((size_t)D_ * D_ * 4);            // v_w transposed
    us*    embb  = (us*)   carve((size_t)B_ * UM1 * D_ * 2);      // embedded tokens
    us*    xatt  = (us*)   carve((size_t)B_ * 768 * 2);           // [e_t|ctx|h] bf16
    float* h_att = (float*)carve((size_t)B_ * D_ * 4);
    float* c_att = (float*)carve((size_t)B_ * D_ * 4);
    float* cdec  = (float*)carve((size_t)B_ * D_ * 4);
    us*    hdec  = (us*)   carve((size_t)2 * B_ * D_ * 2);        // ping-pong dec h
    us*    lstb  = (us*)   carve((size_t)B_ * UM1 * D_ * 2);      // masked att h
    us*    ctxb  = (us*)   carve((size_t)B_ * UM1 * D_ * 2);      // masked ctx
    us*    decb  = (us*)   carve((size_t)B_ * UM1 * D_ * 2);      // masked dec h

    const long ldrow = (long)UM1 * D_;  // 25344
    auto g1 = [](int n) { return dim3((unsigned)((n + 255) / 256)); };

    // ---- prep ----
    k_f2bf<<<g1(B_ * T_ * D_), 256, 0, stream>>>(eout, eoutb, B_ * T_ * D_);
    k_f2bf<<<g1(D_ * D_), 256, 0, stream>>>(w_w, wwb, D_ * D_);
    k_f2bf<<<g1(V_ * 768), 256, 0, stream>>>(cls_w, clsW, V_ * 768);
    k_catw<<<g1(1024 * 768), 256, 0, stream>>>(att_Wih, att_Whh, attW);
    k_catw<<<g1(1024 * 768), 256, 0, stream>>>(dec_Wih, dec_Whh, decW);
    k_trvw<<<g1(D_ * D_), 256, 0, stream>>>(v_w, vwT);
    k_gemb<<<g1(B_ * UM1 * D_), 256, 0, stream>>>(embedding, y, embb);
    k_zero_f<<<g1(B_ * D_), 256, 0, stream>>>(c_att, B_ * D_);
    k_zero_f<<<g1(B_ * D_), 256, 0, stream>>>(cdec, B_ * D_);
    k_zero_us<<<g1(2 * B_ * D_), 256, 0, stream>>>(hdec, 2 * B_ * D_);
    k_initx<<<g1(B_ * 768), 256, 0, stream>>>(xatt, embb);

    // ---- att_h GEMM ----
    k_atth<<<dim3(B_ * T_ / 16, 4), 32, 0, stream>>>(eoutb, wwb, w_b, atthb);

    // ---- attention scan: 99 sequential steps ----
    for (int s = 0; s < UM1; ++s) {
        k_lstm<<<dim3(B_ / 16, D_ / 16), 384, 0, stream>>>(
            xatt, 768, xatt + 256, 768, xatt + 512, 768,
            attW, att_bih, att_bhh,
            c_att, h_att, (us*)nullptr, 0,
            lstb + (long)s * D_, ldrow, ymask, s);
        k_attn<<<B_, 256, 0, stream>>>(
            atthb, eoutb, embb, h_att, vwT, v_b, w_att_v,
            xmask, ymask, xatt, ctxb, s);
    }

    // ---- decoder scan: 99 sequential steps ----
    for (int s = 0; s < UM1; ++s) {
        k_lstm<<<dim3(B_ / 16, D_ / 16), 384, 0, stream>>>(
            lstb + (long)s * D_, ldrow,
            ctxb + (long)s * D_, ldrow,
            hdec + (size_t)(s & 1) * B_ * D_, D_,
            decW, dec_bih, dec_bhh,
            cdec, (float*)nullptr,
            hdec + (size_t)((s + 1) & 1) * B_ * D_, D_,
            decb + (long)s * D_, ldrow, ymask, s);
    }

    // ---- classifier GEMM: (6336 x 5000, K=768), 32x64 per wave ----
    k_cls<<<dim3(B_ * UM1 / 32, (V_ + 63) / 64), 32, 0, stream>>>(
        lstb, ctxb, decb, clsW, cls_b, out);
}
